// PointDGMambaStudent_78323023610183
// MI455X (gfx1250) — compile-verified, matrix operand order checked by reading, not run on hardware
//
#include <hip/hip_runtime.h>
#include <hip/hip_bf16.h>

// ---------------- problem constants ----------------
#define B_SZ   8
#define L_SEQ  1024
#define DM     384           // d_model
#define DIN    768           // d_inner
#define NST    16            // d_state
#define DTR    48            // dt_rank
#define DTRP   64            // dt_rank padded to WMMA K multiple
#define XPN    80            // dt_rank + 2*d_state
#define ROWS   (B_SZ * L_SEQ)   // 8192

// LDS row padding (halves) to avoid bank conflicts on ds_load_b128
#define PADH   8

typedef _Float16 half_t;
typedef __attribute__((ext_vector_type(16))) _Float16 v16h;
typedef __attribute__((ext_vector_type(8)))  _Float16 v8h;
typedef __attribute__((ext_vector_type(8)))  float    v8f;

// ---------------------------------------------------------------------------
// WMMA fragment loader: row-major (R x K) matrix, 16-bit elements.
// CDNA5 16-bit A-matrix 16x32 layout (CDNA5 ISA §7.12.2):
//   lanes 0-15  : halves 0..7 = K k0+0..7,  halves 8..15 = K k0+16..23
//   lanes 16-31 : halves 0..7 = K k0+8..15, halves 8..15 = K k0+24..31
// B operand uses the same pattern with N on the lane dim (weights stored (N,K)).
// Works for both global and LDS-resident tiles (ld = row stride in halves).
// ---------------------------------------------------------------------------
__device__ __forceinline__ v16h load_frag16(const half_t* __restrict__ base,
                                            int row, int ld, int k0, int lane) {
    const half_t* p = base + (size_t)row * ld + k0 + ((lane >> 4) & 1) * 8;
    v8h lo = *reinterpret_cast<const v8h*>(p);
    v8h hi = *reinterpret_cast<const v8h*>(p + 16);
    v16h r;
#pragma unroll
    for (int i = 0; i < 8; ++i) { r[i] = lo[i]; r[8 + i] = hi[i]; }
    return r;
}

// ---------------------------------------------------------------------------
// Async stage of a contiguous global block (nrows x row_bytes) into LDS with a
// padded row stride, using GLOBAL_LOAD_ASYNC_TO_LDS_B128 (ASYNCcnt-tracked).
// Caller must __syncthreads() after this returns.
// ---------------------------------------------------------------------------
__device__ __forceinline__ void async_stage_tile(const half_t* __restrict__ gsrc,
                                                 half_t* sdst,
                                                 int row_bytes, int nrows,
                                                 int lds_row_bytes,
                                                 int tid, int nthreads) {
    const int chunks_per_row = row_bytes >> 4;          // 16B chunks
    const int total = chunks_per_row * nrows;
    for (int c = tid; c < total; c += nthreads) {
        int r  = c / chunks_per_row;
        int cc = c - r * chunks_per_row;
        unsigned int lds_off =
            (unsigned int)(size_t)((const char*)sdst + r * lds_row_bytes + cc * 16);
        unsigned long long gaddr =
            (unsigned long long)(size_t)((const char*)gsrc + (size_t)r * row_bytes + cc * 16);
        asm volatile("global_load_async_to_lds_b128 %0, %1, off"
                     :: "v"(lds_off), "v"(gaddr) : "memory");
    }
    asm volatile("s_wait_asynccnt 0x0" ::: "memory");
}

__device__ __forceinline__ float silu_f(float x) {
    return x / (1.0f + __expf(-x));
}

// ---------------- f32 -> f16 conversion helpers ----------------
__global__ void k_cvt_f16(const float* __restrict__ src, half_t* __restrict__ dst, int n) {
    int i = blockIdx.x * blockDim.x + threadIdx.x;
    if (i < n) dst[i] = (half_t)src[i];
}

// dt_proj_w: (DIN x DTR) -> (DIN x DTRP) zero-padded
__global__ void k_pad_dtw(const float* __restrict__ src, half_t* __restrict__ dst) {
    int i = blockIdx.x * blockDim.x + threadIdx.x;  // over DIN*DTRP
    if (i >= DIN * DTRP) return;
    int r = i / DTRP, c = i % DTRP;
    dst[i] = (c < DTR) ? (half_t)src[r * DTR + c] : (half_t)0.0f;
}

// zero pad columns 48..63 of dlow (ROWS x DTRP)
__global__ void k_zero_dlow_pad(half_t* __restrict__ dlow) {
    int i = blockIdx.x * blockDim.x + threadIdx.x;  // over ROWS*16
    if (i >= ROWS * (DTRP - DTR)) return;
    int r = i / (DTRP - DTR), c = DTR + i % (DTRP - DTR);
    dlow[r * DTRP + c] = (half_t)0.0f;
}

// ---------------- LayerNorm: x (ROWS x 384) -> xn f16 ----------------
__global__ void k_layernorm(const float* __restrict__ x,
                            const float* __restrict__ g,
                            const float* __restrict__ b,
                            half_t* __restrict__ xn) {
    const int row = blockIdx.x;                 // 0..8191
    const int tid = threadIdx.x;                // 0..127
    const float* xr = x + (size_t)row * DM;
    float v[3], s = 0.f, s2 = 0.f;
#pragma unroll
    for (int i = 0; i < 3; ++i) {
        v[i] = xr[tid + 128 * i];
        s += v[i]; s2 += v[i] * v[i];
    }
#pragma unroll
    for (int off = 16; off; off >>= 1) {
        s  += __shfl_xor(s,  off, 32);
        s2 += __shfl_xor(s2, off, 32);
    }
    __shared__ float shs[4], shs2[4];
    if ((tid & 31) == 0) { shs[tid >> 5] = s; shs2[tid >> 5] = s2; }
    __syncthreads();
    float ts  = shs[0] + shs[1] + shs[2] + shs[3];
    float ts2 = shs2[0] + shs2[1] + shs2[2] + shs2[3];
    float mu  = ts * (1.0f / DM);
    float var = ts2 * (1.0f / DM) - mu * mu;
    float rst = rsqrtf(var + 1e-5f);
#pragma unroll
    for (int i = 0; i < 3; ++i) {
        int c = tid + 128 * i;
        xn[(size_t)row * DM + c] = (half_t)((v[i] - mu) * rst * g[c] + b[c]);
    }
}

// ---------------- GEMM1: xz = xn @ in_proj_w^T, split into xh / silu(z) -----
// block = 128 threads (4 waves); A tile (16x384) async-staged in LDS and
// shared by the 4 waves; each wave: 16 rows x 64 cols, K=384.
__global__ void k_gemm_xz(const half_t* __restrict__ Xn,
                          const half_t* __restrict__ W,   // (1536 x 384)
                          float* __restrict__ xh,         // (ROWS x 768)
                          float* __restrict__ sz) {       // silu(z), (ROWS x 768)
    __shared__ half_t sA[16 * (DM + PADH)];
    const int lane = threadIdx.x & 31;
    const int wave = threadIdx.x >> 5;
    const int m0 = blockIdx.x * 16;
    const int n0 = blockIdx.y * 256 + wave * 64;
    const int rloc = lane & 15;

    async_stage_tile(Xn + (size_t)m0 * DM, sA, DM * 2, 16, (DM + PADH) * 2,
                     threadIdx.x, 128);
    __syncthreads();

    v8f acc[4] = {};
    for (int k0 = 0; k0 < DM; k0 += 32) {
        v16h a = load_frag16(sA, rloc, DM + PADH, k0, lane);   // ds_load
        v16h b0 = load_frag16(W, n0 +  0 + rloc, DM, k0, lane);
        v16h b1 = load_frag16(W, n0 + 16 + rloc, DM, k0, lane);
        v16h b2 = load_frag16(W, n0 + 32 + rloc, DM, k0, lane);
        v16h b3 = load_frag16(W, n0 + 48 + rloc, DM, k0, lane);
        acc[0] = __builtin_amdgcn_wmma_f32_16x16x32_f16(false, a, false, b0, (short)0, acc[0], false, false);
        acc[1] = __builtin_amdgcn_wmma_f32_16x16x32_f16(false, a, false, b1, (short)0, acc[1], false, false);
        acc[2] = __builtin_amdgcn_wmma_f32_16x16x32_f16(false, a, false, b2, (short)0, acc[2], false, false);
        acc[3] = __builtin_amdgcn_wmma_f32_16x16x32_f16(false, a, false, b3, (short)0, acc[3], false, false);
    }
    const int mbase = m0 + ((lane >> 4) & 1) * 8;
#pragma unroll
    for (int t = 0; t < 4; ++t) {
        int n = n0 + 16 * t + rloc;
#pragma unroll
        for (int r = 0; r < 8; ++r) {
            float val = acc[t][r];
            size_t row = (size_t)(mbase + r);
            if (n < DIN) xh[row * DIN + n] = val;
            else         sz[row * DIN + (n - DIN)] = silu_f(val);
        }
    }
}

// ---------------- causal depthwise conv1d (width 4) + SiLU ----------------
__global__ void k_conv_silu(const float* __restrict__ xh,
                            const float* __restrict__ cw,  // (768,1,4)
                            const float* __restrict__ cb,
                            float* __restrict__ u,
                            half_t* __restrict__ uh) {
    int idx = blockIdx.x * blockDim.x + threadIdx.x;   // over ROWS*DIN
    if (idx >= ROWS * DIN) return;
    int d   = idx % DIN;
    int row = idx / DIN;          // b*L + l
    int l   = row % L_SEQ;
    const float* w = cw + d * 4;
    float acc = cb[d];
#pragma unroll
    for (int j = 0; j < 4; ++j) {
        int ll = l - 3 + j;
        if (ll >= 0) acc += xh[(size_t)(row - 3 + j) * DIN + d] * w[j];
    }
    float s = silu_f(acc);
    u[idx]  = s;
    uh[idx] = (half_t)s;
}

// ---------------- GEMM2: xpo = u @ x_proj_w^T (ROWS x 80) ----------------
// block = 160 threads (5 waves); A tile (16x768) async-staged in LDS;
// wave w computes 16x16 tile at n0 = 16w. Also emits dlow (f16, K-padded).
__global__ void k_gemm_xpo(const half_t* __restrict__ U,
                           const half_t* __restrict__ Wx,  // (80 x 768)
                           float* __restrict__ xpo,        // (ROWS x 80)
                           half_t* __restrict__ dlow) {    // (ROWS x 64)
    __shared__ half_t sA[16 * (DIN + PADH)];
    const int lane = threadIdx.x & 31;
    const int wave = threadIdx.x >> 5;
    const int m0 = blockIdx.x * 16;
    const int n0 = wave * 16;
    const int rloc = lane & 15;

    async_stage_tile(U + (size_t)m0 * DIN, sA, DIN * 2, 16, (DIN + PADH) * 2,
                     threadIdx.x, 160);
    __syncthreads();

    v8f acc = {};
    for (int k0 = 0; k0 < DIN; k0 += 32) {
        v16h a  = load_frag16(sA, rloc, DIN + PADH, k0, lane);
        v16h bf = load_frag16(Wx, n0 + rloc, DIN, k0, lane);
        acc = __builtin_amdgcn_wmma_f32_16x16x32_f16(
            false, a, false, bf, (short)0, acc, false, false);
    }
    const int n     = n0 + rloc;
    const int mbase = m0 + ((lane >> 4) & 1) * 8;
#pragma unroll
    for (int r = 0; r < 8; ++r) {
        float val = acc[r];
        size_t row = (size_t)(mbase + r);
        xpo[row * XPN + n] = val;
        if (n < DTR) dlow[row * DTRP + n] = (half_t)val;
    }
}

// ---------------- GEMM3: delta = softplus(dlow @ dt_proj_w^T + b) ----------
// block = 384 threads (12 waves); A tile (16x64) async-staged in LDS;
// wave w: 16x64 tile at n0 = 64w; K = 64 (padded).
__global__ void k_gemm_delta(const half_t* __restrict__ Dl,   // (ROWS x 64)
                             const half_t* __restrict__ Wd,   // (768 x 64)
                             const float* __restrict__ bias,  // (768)
                             float* __restrict__ delta) {     // (ROWS x 768)
    __shared__ half_t sA[16 * (DTRP + PADH)];
    const int lane = threadIdx.x & 31;
    const int wave = threadIdx.x >> 5;
    const int m0 = blockIdx.x * 16;
    const int n0 = wave * 64;
    const int rloc = lane & 15;

    async_stage_tile(Dl + (size_t)m0 * DTRP, sA, DTRP * 2, 16, (DTRP + PADH) * 2,
                     threadIdx.x, 384);
    __syncthreads();

    v8f acc[4] = {};
    for (int k0 = 0; k0 < DTRP; k0 += 32) {
        v16h a  = load_frag16(sA, rloc, DTRP + PADH, k0, lane);
        v16h b0 = load_frag16(Wd, n0 +  0 + rloc, DTRP, k0, lane);
        v16h b1 = load_frag16(Wd, n0 + 16 + rloc, DTRP, k0, lane);
        v16h b2 = load_frag16(Wd, n0 + 32 + rloc, DTRP, k0, lane);
        v16h b3 = load_frag16(Wd, n0 + 48 + rloc, DTRP, k0, lane);
        acc[0] = __builtin_amdgcn_wmma_f32_16x16x32_f16(false, a, false, b0, (short)0, acc[0], false, false);
        acc[1] = __builtin_amdgcn_wmma_f32_16x16x32_f16(false, a, false, b1, (short)0, acc[1], false, false);
        acc[2] = __builtin_amdgcn_wmma_f32_16x16x32_f16(false, a, false, b2, (short)0, acc[2], false, false);
        acc[3] = __builtin_amdgcn_wmma_f32_16x16x32_f16(false, a, false, b3, (short)0, acc[3], false, false);
    }
    const int mbase = m0 + ((lane >> 4) & 1) * 8;
#pragma unroll
    for (int t = 0; t < 4; ++t) {
        int n = n0 + 16 * t + rloc;
        float bn = bias[n];
#pragma unroll
        for (int r = 0; r < 8; ++r) {
            float v = acc[t][r] + bn;
            float sp = (v > 20.0f) ? v : log1pf(__expf(v));
            delta[(size_t)(mbase + r) * DIN + n] = sp;
        }
    }
}

// ---------------- selective scan + D skip + SiLU(z) gating ----------------
// one thread per (b, d) channel: 16-state h lives in registers; loop L=1024.
__global__ void k_scan(const float* __restrict__ delta,
                       const float* __restrict__ u,
                       const float* __restrict__ xpo,
                       const float* __restrict__ A_log,
                       const float* __restrict__ Dp,
                       const float* __restrict__ sz,
                       half_t* __restrict__ yh) {
    int t = blockIdx.x * blockDim.x + threadIdx.x;   // 0 .. B*DIN-1
    if (t >= B_SZ * DIN) return;
    int b = t / DIN, d = t % DIN;
    float a[NST], h[NST];
#pragma unroll
    for (int n = 0; n < NST; ++n) {
        a[n] = -__expf(A_log[d * NST + n]);   // A = -exp(A_log)
        h[n] = 0.0f;
    }
    const float Dd = Dp[d];
    const size_t base  = (size_t)b * L_SEQ * DIN + d;
    const size_t xbase = (size_t)b * L_SEQ * XPN;
    for (int l = 0; l < L_SEQ; ++l) {
        size_t idx = base + (size_t)l * DIN;
        float dt = delta[idx];
        float uu = u[idx];
        float Bv[NST], Cv[NST];
        const float4* p = reinterpret_cast<const float4*>(xpo + xbase + (size_t)l * XPN + DTR);
#pragma unroll
        for (int q = 0; q < 4; ++q) {
            float4 f = p[q];
            Bv[4 * q] = f.x; Bv[4 * q + 1] = f.y; Bv[4 * q + 2] = f.z; Bv[4 * q + 3] = f.w;
        }
#pragma unroll
        for (int q = 0; q < 4; ++q) {
            float4 f = p[4 + q];
            Cv[4 * q] = f.x; Cv[4 * q + 1] = f.y; Cv[4 * q + 2] = f.z; Cv[4 * q + 3] = f.w;
        }
        float du = dt * uu;
        float y = 0.0f;
#pragma unroll
        for (int n = 0; n < NST; ++n) {
            h[n] = __expf(dt * a[n]) * h[n] + du * Bv[n];
            y += h[n] * Cv[n];
        }
        float out = (y + uu * Dd) * sz[idx];
        yh[idx] = (half_t)out;
        if (l + 1 < L_SEQ) {
            __builtin_prefetch(&delta[idx + DIN], 0, 3);
            __builtin_prefetch(&u[idx + DIN], 0, 3);
        }
    }
}

// ---------------- GEMM4: out = residual + y @ out_proj_w^T ----------------
// block = 192 threads (6 waves); A tile (16x768) async-staged in LDS;
// wave w: 16x64 tile at n0 = 64w; K = 768; N = 384.
__global__ void k_gemm_out(const half_t* __restrict__ Yh,   // (ROWS x 768)
                           const half_t* __restrict__ Wo,   // (384 x 768)
                           const float* __restrict__ X,     // residual (ROWS x 384)
                           float* __restrict__ out) {       // (ROWS x 384)
    __shared__ half_t sA[16 * (DIN + PADH)];
    const int lane = threadIdx.x & 31;
    const int wave = threadIdx.x >> 5;
    const int m0 = blockIdx.x * 16;
    const int n0 = wave * 64;
    const int rloc = lane & 15;

    async_stage_tile(Yh + (size_t)m0 * DIN, sA, DIN * 2, 16, (DIN + PADH) * 2,
                     threadIdx.x, 192);
    __syncthreads();

    v8f acc[4] = {};
    for (int k0 = 0; k0 < DIN; k0 += 32) {
        v16h a  = load_frag16(sA, rloc, DIN + PADH, k0, lane);
        v16h b0 = load_frag16(Wo, n0 +  0 + rloc, DIN, k0, lane);
        v16h b1 = load_frag16(Wo, n0 + 16 + rloc, DIN, k0, lane);
        v16h b2 = load_frag16(Wo, n0 + 32 + rloc, DIN, k0, lane);
        v16h b3 = load_frag16(Wo, n0 + 48 + rloc, DIN, k0, lane);
        acc[0] = __builtin_amdgcn_wmma_f32_16x16x32_f16(false, a, false, b0, (short)0, acc[0], false, false);
        acc[1] = __builtin_amdgcn_wmma_f32_16x16x32_f16(false, a, false, b1, (short)0, acc[1], false, false);
        acc[2] = __builtin_amdgcn_wmma_f32_16x16x32_f16(false, a, false, b2, (short)0, acc[2], false, false);
        acc[3] = __builtin_amdgcn_wmma_f32_16x16x32_f16(false, a, false, b3, (short)0, acc[3], false, false);
    }
    const int mbase = m0 + ((lane >> 4) & 1) * 8;
#pragma unroll
    for (int t = 0; t < 4; ++t) {
        int n = n0 + 16 * t + rloc;
#pragma unroll
        for (int r = 0; r < 8; ++r) {
            size_t row = (size_t)(mbase + r);
            out[row * DM + n] = X[row * DM + n] + acc[t][r];
        }
    }
}

// ---------------------------------------------------------------------------
static inline size_t alignup(size_t x) { return (x + 255) & ~(size_t)255; }

extern "C" void kernel_launch(void* const* d_in, const int* in_sizes, int n_in,
                              void* d_out, int out_size, void* d_ws, size_t ws_size,
                              hipStream_t stream) {
    (void)in_sizes; (void)n_in; (void)out_size; (void)ws_size;
    const float* x        = (const float*)d_in[0];
    const float* ln_g     = (const float*)d_in[1];
    const float* ln_b     = (const float*)d_in[2];
    const float* in_w     = (const float*)d_in[3];
    const float* conv_w   = (const float*)d_in[4];
    const float* conv_b   = (const float*)d_in[5];
    const float* xp_w     = (const float*)d_in[6];
    const float* dt_w     = (const float*)d_in[7];
    const float* dt_b     = (const float*)d_in[8];
    const float* A_log    = (const float*)d_in[9];
    const float* D_param  = (const float*)d_in[10];
    const float* out_w    = (const float*)d_in[11];
    float* out = (float*)d_out;

    // ---- workspace carve-out ----
    char* ws = (char*)d_ws;
    size_t off = 0;
    auto take = [&](size_t bytes) { char* p = ws + off; off += alignup(bytes); return p; };

    half_t* xn_h   = (half_t*)take((size_t)ROWS * DM * 2);
    half_t* w1_h   = (half_t*)take((size_t)2 * DIN * DM * 2);      // 1536x384
    half_t* xpw_h  = (half_t*)take((size_t)XPN * DIN * 2);         // 80x768
    half_t* dtw_h  = (half_t*)take((size_t)DIN * DTRP * 2);        // 768x64
    half_t* outw_h = (half_t*)take((size_t)DM * DIN * 2);          // 384x768
    float*  xh     = (float*)take((size_t)ROWS * DIN * 4);         // aliased by delta later
    float*  sz     = (float*)take((size_t)ROWS * DIN * 4);
    float*  u      = (float*)take((size_t)ROWS * DIN * 4);
    half_t* u_h    = (half_t*)take((size_t)ROWS * DIN * 2);
    float*  xpo    = (float*)take((size_t)ROWS * XPN * 4);
    half_t* dlow   = (half_t*)take((size_t)ROWS * DTRP * 2);
    half_t* y_h    = (half_t*)take((size_t)ROWS * DIN * 2);
    float*  delta  = xh;   // xh is dead after the conv kernel -> reuse for delta

    // ---- weight conversions to f16 ----
    {
        int n1 = 2 * DIN * DM;   // 589824
        k_cvt_f16<<<(n1 + 255) / 256, 256, 0, stream>>>(in_w, w1_h, n1);
        int n2 = XPN * DIN;      // 61440
        k_cvt_f16<<<(n2 + 255) / 256, 256, 0, stream>>>(xp_w, xpw_h, n2);
        int n3 = DM * DIN;       // 294912
        k_cvt_f16<<<(n3 + 255) / 256, 256, 0, stream>>>(out_w, outw_h, n3);
        int n4 = DIN * DTRP;     // 49152
        k_pad_dtw<<<(n4 + 255) / 256, 256, 0, stream>>>(dt_w, dtw_h);
        int n5 = ROWS * (DTRP - DTR);
        k_zero_dlow_pad<<<(n5 + 255) / 256, 256, 0, stream>>>(dlow);
    }

    // ---- 1. layernorm -> xn (f16) ----
    k_layernorm<<<ROWS, 128, 0, stream>>>(x, ln_g, ln_b, xn_h);

    // ---- 2. in_proj GEMM -> xh (f32), silu(z) (f32) ----
    {
        dim3 grid(ROWS / 16, (2 * DIN) / 256);   // (512, 6)
        k_gemm_xz<<<grid, 128, 0, stream>>>(xn_h, w1_h, xh, sz);
    }

    // ---- 3. depthwise conv + silu -> u (f32 + f16) ----
    {
        int n = ROWS * DIN;
        k_conv_silu<<<(n + 255) / 256, 256, 0, stream>>>(xh, conv_w, conv_b, u, u_h);
    }

    // ---- 4. x_proj GEMM -> xpo (f32) + dlow (f16, padded) ----
    {
        dim3 grid(ROWS / 16, 1);                 // 5 waves/block cover N=80
        k_gemm_xpo<<<grid, 160, 0, stream>>>(u_h, xpw_h, xpo, dlow);
    }

    // ---- 5. dt_proj GEMM + softplus -> delta (f32, aliases xh) ----
    {
        dim3 grid(ROWS / 16, 1);                 // 12 waves/block cover N=768
        k_gemm_delta<<<grid, 384, 0, stream>>>(dlow, dtw_h, dt_b, delta);
    }

    // ---- 6. selective scan + D skip + gate -> y (f16) ----
    {
        int n = B_SZ * DIN;                      // 6144
        k_scan<<<(n + 255) / 256, 256, 0, stream>>>(delta, u, xpo, A_log, D_param, sz, y_h);
    }

    // ---- 7. out_proj GEMM + residual -> out (f32) ----
    {
        dim3 grid(ROWS / 16, 1);                 // 6 waves/block cover N=384
        k_gemm_out<<<grid, 192, 0, stream>>>(y_h, outw_h, x, out);
    }
}
